// Attention_64707977281620
// MI455X (gfx1250) — matrix operand order repaired and checked by reading, not compiled, not run on hardware
//
#include <hip/hip_runtime.h>

// ---------------------------------------------------------------------------
// CDNA5 (gfx1250) fused gated multi-head attention, bf16 WMMA pipeline with
// GLOBAL_LOAD_ASYNC_TO_LDS staging (ASYNCcnt) for bf16 operand tiles except attention's Q tile.
// Shapes: B=1, R=128, Q=K=256, C=256, H=8, D=32.
// ---------------------------------------------------------------------------

typedef __attribute__((ext_vector_type(16))) __bf16 bf16x16;
typedef __attribute__((ext_vector_type(8)))  float  f32x8;

__device__ __forceinline__ __bf16 f2bf(float f) {
  unsigned u = __builtin_bit_cast(unsigned, f);
  unsigned r = (u + 0x7fffu + ((u >> 16) & 1u)) >> 16;   // round-nearest-even
  unsigned short s = (unsigned short)r;
  return __builtin_bit_cast(__bf16, s);
}
__device__ __forceinline__ float bf2f(__bf16 b) {
  unsigned short s = __builtin_bit_cast(unsigned short, b);
  unsigned u = ((unsigned)s) << 16;
  return __builtin_bit_cast(float, u);
}

// Wave-relative LDS byte offset of a __shared__ object (generic-pointer low bits).
__device__ __forceinline__ unsigned lds_off(const void* p) {
  return (unsigned)(unsigned long long)p;
}

// 16B global -> LDS async copy (ASYNCcnt). INST_OFFSET applies to BOTH the LDS
// destination and the global source (ISA 10.7 pseudocode), so a 64B copy needs
// one base address + immediate offsets only.
__device__ __forceinline__ void async_b128(unsigned lds_byte_off, const void* g) {
  asm volatile("global_load_async_to_lds_b128 %0, %1, off"
               :: "v"(lds_byte_off), "v"(g) : "memory");
}
__device__ __forceinline__ void async_b128x4(unsigned lds_byte_off, const void* g) {
  asm volatile(
      "global_load_async_to_lds_b128 %0, %1, off\n\t"
      "global_load_async_to_lds_b128 %0, %1, off offset:16\n\t"
      "global_load_async_to_lds_b128 %0, %1, off offset:32\n\t"
      "global_load_async_to_lds_b128 %0, %1, off offset:48"
      :: "v"(lds_byte_off), "v"(g) : "memory");
}
__device__ __forceinline__ void wait_async() {
  asm volatile("s_wait_asynccnt 0" ::: "memory");
}

// A-fragment: 16x32 bf16, row-major source [m][k] with leading dim `ld`.
// ISA 7.12.2: lanes 0-15 -> M=lane, K = {0..7, 16..23}; lanes 16-31 -> K = {8..15, 24..31}.
__device__ __forceinline__ bf16x16 load_frag_a(const __bf16* s, int ld) {
  const int lane = threadIdx.x & 31;
  const __bf16* row = s + (lane & 15) * ld;
  const int kh = (lane >> 4) << 3;                 // 0 or 8
  bf16x16 a;
#pragma unroll
  for (int e = 0; e < 8; ++e) {
    a[e]     = row[kh + e];
    a[8 + e] = row[16 + kh + e];
  }
  return a;
}

// B-fragment: 32x16 bf16 (K x N), sourced from N-major storage [n][k], ld = k stride.
// lanes 0-15 -> N=lane, K = {0..7, 16..23}; lanes 16-31 -> N=lane-16, K = {8..15, 24..31}.
__device__ __forceinline__ bf16x16 load_frag_b(const __bf16* s, int ld) {
  const int lane = threadIdx.x & 31;
  const __bf16* col = s + (lane & 15) * ld + ((lane >> 4) << 3);
  bf16x16 b;
#pragma unroll
  for (int e = 0; e < 8; ++e) {
    b[e]     = col[e];
    b[8 + e] = col[16 + e];
  }
  return b;
}

__device__ __forceinline__ f32x8 wmma_bf16(bf16x16 a, bf16x16 b, f32x8 c) {
  return __builtin_amdgcn_wmma_f32_16x16x32_bf16(false, a, false, b,
                                                 (short)0, c, false, false);
}

// ---------------------------------------------------------------------------
// Kernel 0: transpose + convert 5 weight matrices [k][n] f32 -> [n][k] bf16.
// Order: Wq, Wk, Wv, Wg, Wo.
// ---------------------------------------------------------------------------
__global__ __launch_bounds__(256) void prep_weights(
    const float* __restrict__ Wq, const float* __restrict__ Wk,
    const float* __restrict__ Wv, const float* __restrict__ Wg,
    const float* __restrict__ Wo, __bf16* __restrict__ wt) {
  const int idx = blockIdx.x * 256 + threadIdx.x;   // 5 * 65536 elements
  const int w   = idx >> 16;
  const int rem = idx & 65535;
  const int k = rem >> 8, n = rem & 255;
  const float* src = (w == 0) ? Wq : (w == 1) ? Wk : (w == 2) ? Wv
                   : (w == 3) ? Wg : Wo;
  wt[w * 65536 + n * 256 + k] = f2bf(src[k * 256 + n]);
}

// ---------------------------------------------------------------------------
// Kernel 1: projections. grid = (512 row-strips, 4 outputs).
// which: 0 -> Qp = q_x@Wq (bf16), 1 -> Kp = k_x@Wk, 2 -> Vt = (v_x@Wv)^T,
//        3 -> Gb = sigmoid(q_x@Wg + bg) (bf16).
// Per block: C[64,256] = A[64,256] x B[256,256], K-tiled by 32.
// ---------------------------------------------------------------------------
__global__ __launch_bounds__(256) void proj_kernel(
    const float* __restrict__ qx, const float* __restrict__ kx,
    const float* __restrict__ vx, const __bf16* __restrict__ wt,
    const float* __restrict__ bg,
    __bf16* __restrict__ Qp, __bf16* __restrict__ Kp,
    __bf16* __restrict__ Vt, __bf16* __restrict__ Gb) {
  __shared__ __bf16 Xs[64][40];     // 64x32 A tile (padded, 16B-aligned rows)
  __shared__ __bf16 Ws[256][40];    // 256x32 B tile, N-major (padded)

  const int strip = blockIdx.x;     // 64-row strip of the 32768-row activation
  const int which = blockIdx.y;
  const float*  X = (which == 1) ? kx : (which == 2) ? vx : qx;
  const __bf16* W = wt + which * 65536;

  const int t = threadIdx.x;
  const int wave = t >> 5, lane = t & 31;
  const int rt = wave & 3;          // row tile (4 x 16 rows)
  const int cg = wave >> 2;         // col-tile group (8 tiles of 16 cols)
  const int mbase = (lane >> 4) << 3;
  const int nlane = lane & 15;

  f32x8 acc[8];
#pragma unroll
  for (int i = 0; i < 8; ++i) acc[i] = {};

  for (int ks = 0; ks < 8; ++ks) {
    const int k0 = ks * 32;
    // B tile: bf16 in global -> LDS via async-DMA (no VGPR round trip).
    async_b128x4(lds_off(&Ws[t][0]), &W[t * 256 + k0]);
    // A tile: f32 -> bf16 conversion must go through VGPRs.
#pragma unroll
    for (int i = 0; i < 8; ++i) {
      const int idx = t * 8 + i;
      const int rr = idx >> 5, cc = idx & 31;
      Xs[rr][cc] = f2bf(X[(strip * 64 + rr) * 256 + k0 + cc]);
    }
    if (k0 + 32 < 256)   // near (WGP) prefetch of next activation k-tile
      __builtin_prefetch(&X[(strip * 64 + (t >> 2)) * 256 + k0 + 32], 0, 3);
    wait_async();
    __syncthreads();

    // Software-pipelined fragment loads: B(i+1) issued before WMMA(i).
    bf16x16 a = load_frag_a(&Xs[rt * 16][0], 40);
    bf16x16 b = load_frag_b(&Ws[(cg * 8) * 16][0], 40);
#pragma unroll
    for (int i = 0; i < 8; ++i) {
      bf16x16 bn = b;
      if (i < 7) bn = load_frag_b(&Ws[(cg * 8 + i + 1) * 16][0], 40);
      acc[i] = wmma_bf16(a, b, acc[i]);
      b = bn;
    }
    __syncthreads();
  }

#pragma unroll
  for (int i = 0; i < 8; ++i) {
    const int n = (cg * 8 + i) * 16 + nlane;
#pragma unroll
    for (int e = 0; e < 8; ++e) {
      const int gr = strip * 64 + rt * 16 + mbase + e;  // = r*256 + s
      const float v = acc[i][e];
      if (which == 0) {
        Qp[gr * 256 + n] = f2bf(v);
      } else if (which == 1) {
        Kp[gr * 256 + n] = f2bf(v);
      } else if (which == 2) {
        const int r = gr >> 8, key = gr & 255;
        Vt[(r * 256 + n) * 256 + key] = f2bf(v);        // transposed for P.V B-frags
      } else {
        const float g = 1.0f / (1.0f + __expf(-(v + bg[n])));
        Gb[gr * 256 + n] = f2bf(g);
      }
    }
  }
}

// ---------------------------------------------------------------------------
// Kernel 2: fused attention core. grid = R*H*(Q/32) = 8192 blocks.
// Per block (r, h, qs): S = Q Kt * norm + bias -> softmax -> O = P V -> gate.
// Scores never touch HBM (would cost 2x268 MB); kept in a 32x256 LDS tile.
// ---------------------------------------------------------------------------
__global__ __launch_bounds__(256) void attn_kernel(
    const __bf16* __restrict__ Qp, const __bf16* __restrict__ Kp,
    const __bf16* __restrict__ Vt, const __bf16* __restrict__ Gb,
    const float* __restrict__ bias, __bf16* __restrict__ Og) {
  __shared__ __bf16 Qs[32][40];     // 32x32 query tile
  __shared__ __bf16 Ks[256][40];    // all 256 keys x 32 dims (N-major for B-frags)
  __shared__ __bf16 Vs[32][264];    // V^T: 32 dims x 256 keys (N-major for B-frags)
  __shared__ __bf16 Sb[32][264];    // scores -> exp -> normalized P, in place
  __shared__ float  red[32][8];
  __shared__ float  rowmax[32];
  __shared__ float  rowinv[32];

  const int b  = blockIdx.x;
  const int qs = b & 7;
  const int h  = (b >> 3) & 7;
  const int r  = b >> 6;

  const int t = threadIdx.x;
  const int wave = t >> 5, lane = t & 31;
  const int mbase = (lane >> 4) << 3;
  const int nlane = lane & 15;
  const float kNorm = 0.17677669529663687f;   // 1/sqrt(32)

  // ---- stage K (256x32) and V^T (32x256) via async-DMA, Q via VGPRs ----
  async_b128x4(lds_off(&Ks[t][0]), &Kp[(r * 256 + t) * 256 + h * 32]);
  {
    const int d = t >> 3, seg = (t & 7) * 32;
    async_b128x4(lds_off(&Vs[d][seg]),
                 &Vt[(r * 256 + h * 32 + d) * 256 + seg]);
  }
#pragma unroll
  for (int i = 0; i < 4; ++i) {               // stage Q (small)
    const int idx = t * 4 + i;
    const int rr = idx >> 5, cc = idx & 31;
    Qs[rr][cc] = Qp[(r * 256 + qs * 32 + rr) * 256 + h * 32 + cc];
  }
  wait_async();
  __syncthreads();

  // ---- scores: 32x256, single K=32 WMMA per 16x16 tile (D == 32) ----
  {
    const int rt = wave >> 2;                 // 0..1 row tile
    const int c4 = wave & 3;                  // group of 4 col tiles
    bf16x16 a = load_frag_a(&Qs[rt * 16][0], 40);
    bf16x16 bb = load_frag_b(&Ks[(c4 * 4) * 16][0], 40);
    f32x8 sacc[4];
#pragma unroll
    for (int i = 0; i < 4; ++i) {
      bf16x16 bn = bb;
      if (i < 3) bn = load_frag_b(&Ks[(c4 * 4 + i + 1) * 16][0], 40);
      f32x8 s = {};
      s = wmma_bf16(a, bb, s);
      sacc[i] = s;
      bb = bn;
    }
#pragma unroll
    for (int i = 0; i < 4; ++i) {
      const int n = (c4 * 4 + i) * 16 + nlane;
#pragma unroll
      for (int e = 0; e < 8; ++e) {
        const int row = rt * 16 + mbase + e;
        const int q = qs * 32 + row;
        const float sv = sacc[i][e] * kNorm + bias[(h * 256 + q) * 256 + n];
        Sb[row][n] = f2bf(sv);
      }
    }
  }
  __syncthreads();

  // ---- softmax: 8 threads per row, in-place on Sb ----
  {
    const int row = t & 31;
    const int seg = t >> 5;
    const int c0 = seg * 32;
    float m = -1e30f;
#pragma unroll 8
    for (int c = 0; c < 32; ++c) m = fmaxf(m, bf2f(Sb[row][c0 + c]));
    red[row][seg] = m;
    __syncthreads();
    if (t < 32) {
      float mm = red[t][0];
#pragma unroll
      for (int s2 = 1; s2 < 8; ++s2) mm = fmaxf(mm, red[t][s2]);
      rowmax[t] = mm;
    }
    __syncthreads();
    const float rmax = rowmax[row];
    float sum = 0.f;
#pragma unroll 8
    for (int c = 0; c < 32; ++c) {
      const float e = __expf(bf2f(Sb[row][c0 + c]) - rmax);
      Sb[row][c0 + c] = f2bf(e);
      sum += e;
    }
    red[row][seg] = sum;
    __syncthreads();
    if (t < 32) {
      float ss = red[t][0];
#pragma unroll
      for (int s2 = 1; s2 < 8; ++s2) ss += red[t][s2];
      rowinv[t] = 1.0f / ss;
    }
    __syncthreads();
    const float inv = rowinv[row];
#pragma unroll 8
    for (int c = 0; c < 32; ++c)
      Sb[row][c0 + c] = f2bf(bf2f(Sb[row][c0 + c]) * inv);
  }
  __syncthreads();

  // ---- O = P V (32x32, K=256), gate, store (waves 0..3; wave-uniform) ----
  if (wave < 4) {
    const int rt = wave >> 1, ct = wave & 1;
    f32x8 acc = {};
    bf16x16 a  = load_frag_a(&Sb[rt * 16][0], 264);
    bf16x16 bb = load_frag_b(&Vs[ct * 16][0], 264);
#pragma unroll
    for (int ks = 0; ks < 8; ++ks) {
      bf16x16 an = a, bn = bb;
      if (ks < 7) {
        an = load_frag_a(&Sb[rt * 16][(ks + 1) * 32], 264);
        bn = load_frag_b(&Vs[ct * 16][(ks + 1) * 32], 264);
      }
      acc = wmma_bf16(a, bb, acc);
      a = an; bb = bn;
    }
    const int d = ct * 16 + nlane;
#pragma unroll
    for (int e = 0; e < 8; ++e) {
      const int row = rt * 16 + mbase + e;
      const int q = qs * 32 + row;
      const int gi = (r * 256 + q) * 256 + h * 32 + d;
      Og[gi] = f2bf(acc[e] * bf2f(Gb[gi]));
    }
  }
}

// ---------------------------------------------------------------------------
// Kernel 3: output projection out = Og @ Wo + bo, f32 result.
// Both operand tiles are bf16 in global -> staged with async-DMA.
// ---------------------------------------------------------------------------
__global__ __launch_bounds__(256) void outproj_kernel(
    const __bf16* __restrict__ Og, const __bf16* __restrict__ Wot,
    const float* __restrict__ bo, float* __restrict__ out) {
  __shared__ __bf16 As[64][40];
  __shared__ __bf16 Ws[256][40];

  const int strip = blockIdx.x;
  const int t = threadIdx.x;
  const int wave = t >> 5, lane = t & 31;
  const int rt = wave & 3, cg = wave >> 2;
  const int mbase = (lane >> 4) << 3, nlane = lane & 15;

  f32x8 acc[8];
#pragma unroll
  for (int i = 0; i < 8; ++i) acc[i] = {};

  for (int ks = 0; ks < 8; ++ks) {
    const int k0 = ks * 32;
    {
      const int rr = t >> 2, cc = (t & 3) * 8;          // one b128 per thread
      async_b128(lds_off(&As[rr][cc]),
                 &Og[(strip * 64 + rr) * 256 + k0 + cc]);
      async_b128x4(lds_off(&Ws[t][0]), &Wot[t * 256 + k0]);
    }
    wait_async();
    __syncthreads();

    bf16x16 a = load_frag_a(&As[rt * 16][0], 40);
    bf16x16 b = load_frag_b(&Ws[(cg * 8) * 16][0], 40);
#pragma unroll
    for (int i = 0; i < 8; ++i) {
      bf16x16 bn = b;
      if (i < 7) bn = load_frag_b(&Ws[(cg * 8 + i + 1) * 16][0], 40);
      acc[i] = wmma_bf16(a, b, acc[i]);
      b = bn;
    }
    __syncthreads();
  }

#pragma unroll
  for (int i = 0; i < 8; ++i) {
    const int n = (cg * 8 + i) * 16 + nlane;
#pragma unroll
    for (int e = 0; e < 8; ++e) {
      const int gr = strip * 64 + rt * 16 + mbase + e;
      out[gr * 256 + n] = acc[i][e] + bo[n];
    }
  }
}

// ---------------------------------------------------------------------------
// Host entry point.
// Inputs: 0 q_x, 1 k_x, 2 v_x, 3 bias, 4 Wq, 5 Wk, 6 Wv, 7 Wo, 8 bo, 9 Wg, 10 bg
// ---------------------------------------------------------------------------
extern "C" void kernel_launch(void* const* d_in, const int* in_sizes, int n_in,
                              void* d_out, int out_size, void* d_ws, size_t ws_size,
                              hipStream_t stream) {
  const float* qx   = (const float*)d_in[0];
  const float* kx   = (const float*)d_in[1];
  const float* vx   = (const float*)d_in[2];
  const float* bias = (const float*)d_in[3];
  const float* Wq   = (const float*)d_in[4];
  const float* Wk   = (const float*)d_in[5];
  const float* Wv   = (const float*)d_in[6];
  const float* Wo   = (const float*)d_in[7];
  const float* bo   = (const float*)d_in[8];
  const float* Wg   = (const float*)d_in[9];
  const float* bg   = (const float*)d_in[10];
  float* out = (float*)d_out;
  (void)in_sizes; (void)n_in; (void)out_size; (void)ws_size;

  char* ws = (char*)d_ws;
  size_t off = 0;
  auto take = [&](size_t bytes) {
    char* p = ws + off;
    off += (bytes + 255) & ~(size_t)255;
    return p;
  };
  __bf16* wt = (__bf16*)take((size_t)5 * 65536 * 2);       // bf16 weights [n][k]
  __bf16* Qp = (__bf16*)take((size_t)32768 * 256 * 2);     // projected Q (bf16)
  __bf16* Kp = (__bf16*)take((size_t)32768 * 256 * 2);     // projected K (bf16)
  __bf16* Vt = (__bf16*)take((size_t)32768 * 256 * 2);     // projected V, transposed
  __bf16* Gb = (__bf16*)take((size_t)32768 * 256 * 2);     // sigmoid gate (bf16)
  __bf16* Og = (__bf16*)take((size_t)32768 * 256 * 2);     // gated attention output

  prep_weights<<<1280, 256, 0, stream>>>(Wq, Wk, Wv, Wg, Wo, wt);
  proj_kernel<<<dim3(512, 4), 256, 0, stream>>>(qx, kx, vx, wt, bg,
                                                Qp, Kp, Vt, Gb);
  attn_kernel<<<8192, 256, 0, stream>>>(Qp, Kp, Vt, Gb, bias, Og);
  outproj_kernel<<<512, 256, 0, stream>>>(Og, wt + 4 * 65536, bo, out);
}
